// CustomAttention_5222680232329
// MI455X (gfx1250) — matrix operand order repaired and checked by reading, not compiled, not run on hardware
//
#include <hip/hip_runtime.h>
#include <hip/hip_bf16.h>
#include <math.h>

typedef __bf16 bf16_t;
typedef __attribute__((ext_vector_type(16))) __bf16 v16bf;
typedef __attribute__((ext_vector_type(8)))  __bf16 bf16x8;
typedef __attribute__((ext_vector_type(4)))  __bf16 bf16x4;
typedef __attribute__((ext_vector_type(8)))  float  v8f;
typedef __attribute__((ext_vector_type(4)))  float  f32x4;

#define LOGIT_SCALE_MAX 4.6051701860f   // log(100)

static __device__ __forceinline__ v8f wmma_bf16(v16bf a, v16bf b, v8f c) {
    return __builtin_amdgcn_wmma_f32_16x16x32_bf16(false, a, false, b, (short)0, c, false, false);
}

// A-fragment from a bf16 row (16 contiguous from 2*baseSel).
static __device__ __forceinline__ v16bf afrag_bf16(const bf16_t* row, int baseSel) {
    bf16x8 lo = *(const bf16x8*)(row + 2 * baseSel);
    bf16x8 hi = *(const bf16x8*)(row + 2 * baseSel + 8);
    v16bf r;
#pragma unroll
    for (int i = 0; i < 8; ++i) { r[i] = lo[i]; r[8 + i] = hi[i]; }
    return r;
}

// B-fragment from 16 contiguous bf16 (caller folds lane-half k offset).
static __device__ __forceinline__ v16bf bfrag_bf16(const bf16_t* p) {
    bf16x8 lo = *(const bf16x8*)p;
    bf16x8 hi = *(const bf16x8*)(p + 8);
    v16bf r;
#pragma unroll
    for (int i = 0; i < 8; ++i) { r[i] = lo[i]; r[8 + i] = hi[i]; }
    return r;
}

// ---------------------------------------------------------------------------
// Bulk fp32 -> bf16 conversion (one pass over inputs/weights, 4 elems/thread).
// ---------------------------------------------------------------------------
__global__ __launch_bounds__(256) void cvt_f32_bf16_kernel(
    const float* __restrict__ src, bf16_t* __restrict__ dst)
{
    const int i = (blockIdx.x * 256 + threadIdx.x) * 4;
    f32x4 v = *(const f32x4*)(src + i);
    bf16x4 o;
#pragma unroll
    for (int e = 0; e < 4; ++e) o[e] = (bf16_t)v[e];
    *(bf16x4*)(dst + i) = o;
}

// ---------------------------------------------------------------------------
// QKV projection: Out = X @ W_z^T + b_z, stored bf16 head-major [b*16+h][n][d]
// X rows are flattened (n, b) with row = n*4 + b.  M=4096, N=1024, K=1024.
// Block = 8 waves (4 row x 2 col), each wave a 32x64 tile -> block 128x128.
// ---------------------------------------------------------------------------
__global__ __launch_bounds__(256) void qkv_proj_kernel(
    const bf16_t* __restrict__ qc, const bf16_t* __restrict__ kc,
    const bf16_t* __restrict__ vc, const bf16_t* __restrict__ w,
    const float* __restrict__ bias, bf16_t* __restrict__ qb,
    bf16_t* __restrict__ kb, bf16_t* __restrict__ vb)
{
    const int z = blockIdx.z;
    const bf16_t* A  = (z == 0) ? qc : (z == 1) ? kc : vc;
    const bf16_t* W  = w + (size_t)z * 1024 * 1024;
    const float*  bi = bias + z * 1024;
    bf16_t* Out      = (z == 0) ? qb : (z == 1) ? kb : vb;

    const int lane = threadIdx.x, wv = threadIdx.y;
    const int laneRow = lane & 15, half = lane >> 4;
    const int baseSel = half * 8, koff = half * 16;
    const int rBase = blockIdx.x * 128 + (wv & 3) * 32;
    const int nBase = blockIdx.y * 128 + (wv >> 2) * 64;

    v8f acc[2][4];
#pragma unroll
    for (int tr = 0; tr < 2; ++tr)
#pragma unroll
        for (int tc = 0; tc < 4; ++tc) acc[tr][tc] = {};

    const bf16_t* a0p = A + (size_t)(rBase + laneRow) * 1024;
    const bf16_t* a1p = A + (size_t)(rBase + 16 + laneRow) * 1024;
    const bf16_t* bp[4];
#pragma unroll
    for (int t = 0; t < 4; ++t)
        bp[t] = W + (size_t)(nBase + t * 16 + laneRow) * 1024 + koff;

    for (int kk = 0; kk < 1024; kk += 32) {
        v16bf a0 = afrag_bf16(a0p + kk, baseSel);
        v16bf a1 = afrag_bf16(a1p + kk, baseSel);
#pragma unroll
        for (int t = 0; t < 4; ++t) {
            v16bf b = bfrag_bf16(bp[t] + kk);
            acc[0][t] = wmma_bf16(a0, b, acc[0][t]);
            acc[1][t] = wmma_bf16(a1, b, acc[1][t]);
        }
    }

#pragma unroll
    for (int tr = 0; tr < 2; ++tr)
#pragma unroll
        for (int tc = 0; tc < 4; ++tc)
#pragma unroll
            for (int i = 0; i < 8; ++i) {
                int r = rBase + tr * 16 + (half ? i + 8 : i);
                int c = nBase + tc * 16 + laneRow;
                float v = acc[tr][tc][i] + bi[c];
                int n = r >> 2, b = r & 3, h = c >> 6, d = c & 63;
                Out[(((size_t)(b * 16 + h)) * 1024 + n) * 64 + d] = (bf16_t)v;
            }
}

// ---------------------------------------------------------------------------
// In-place L2 normalization over hd=64 of q and k (head-major rows).
// ---------------------------------------------------------------------------
__global__ __launch_bounds__(32) void l2norm_kernel(bf16_t* __restrict__ qb,
                                                    bf16_t* __restrict__ kb)
{
    bf16_t* base = blockIdx.z ? kb : qb;
    bf16_t* row  = base + ((size_t)blockIdx.y * 1024 + blockIdx.x) * 64;
    const int lane = threadIdx.x;
    float v0 = (float)row[lane * 2];
    float v1 = (float)row[lane * 2 + 1];
    float s = v0 * v0 + v1 * v1;
#pragma unroll
    for (int off = 16; off >= 1; off >>= 1) s += __shfl_xor(s, off);
    float inv = 1.0f / fmaxf(sqrtf(s), 1e-12f);
    row[lane * 2]     = (bf16_t)(v0 * inv);
    row[lane * 2 + 1] = (bf16_t)(v1 * inv);
}

// ---------------------------------------------------------------------------
// Flash-style attention, one wave per (head, 16-query block).
// ---------------------------------------------------------------------------
__global__ __launch_bounds__(32) void attn_kernel(
    const bf16_t* __restrict__ qn, const bf16_t* __restrict__ kn,
    const bf16_t* __restrict__ vb, const float* __restrict__ logit_scale,
    bf16_t* __restrict__ x)
{
    __shared__ __align__(16) bf16_t vt[64][32];    // V^T tile: [d][key]
    __shared__ __align__(16) bf16_t pbuf[16][32];  // P repack buffer

    const int qblk = blockIdx.x;
    const int hb   = blockIdx.y;            // b*16 + h
    const int b    = hb >> 4, h = hb & 15;
    const int lane = threadIdx.x;
    const int laneRow = lane & 15, half = lane >> 4, baseSel = half * 8;

    const float ls = __expf(fminf(logit_scale[h], LOGIT_SCALE_MAX));

    const bf16_t* qrow = qn + ((size_t)hb * 1024 + qblk * 16 + laneRow) * 64;
    const v16bf aq0 = afrag_bf16(qrow, baseSel);
    const v16bf aq1 = afrag_bf16(qrow + 32, baseSel);

    float m[8], l[8];
#pragma unroll
    for (int i = 0; i < 8; ++i) { m[i] = -INFINITY; l[i] = 0.0f; }
    v8f o0 = {}, o1 = {}, o2 = {}, o3 = {};

    for (int j = 0; j < 32; ++j) {
        const int keyBase = j * 32;

        const bf16_t* vrow = vb + ((size_t)hb * 1024 + keyBase + lane) * 64;
        bf16x8 vv[8];
#pragma unroll
        for (int t = 0; t < 8; ++t) vv[t] = ((const bf16x8*)vrow)[t];

        __syncthreads();
#pragma unroll
        for (int t = 0; t < 8; ++t)
#pragma unroll
            for (int e = 0; e < 8; ++e)
                vt[t * 8 + e][lane] = vv[t][e];

        const bf16_t* krow0 = kn + ((size_t)hb * 1024 + keyBase + laneRow) * 64 + half * 16;
        const bf16_t* krow1 = kn + ((size_t)hb * 1024 + keyBase + 16 + laneRow) * 64 + half * 16;
        v8f s0 = {}, s1 = {};
        s0 = wmma_bf16(aq0, bfrag_bf16(krow0), s0);
        s0 = wmma_bf16(aq1, bfrag_bf16(krow0 + 32), s0);
        s1 = wmma_bf16(aq0, bfrag_bf16(krow1), s1);
        s1 = wmma_bf16(aq1, bfrag_bf16(krow1 + 32), s1);

        float alpha[8];
#pragma unroll
        for (int i = 0; i < 8; ++i) {
            float p0 = s0[i] * ls, p1 = s1[i] * ls;
            float mx = fmaxf(p0, p1);
#pragma unroll
            for (int off = 8; off >= 1; off >>= 1) mx = fmaxf(mx, __shfl_xor(mx, off));
            float mnew = fmaxf(m[i], mx);
            alpha[i] = __expf(m[i] - mnew);
            p0 = __expf(p0 - mnew);
            p1 = __expf(p1 - mnew);
            float rs = p0 + p1;
#pragma unroll
            for (int off = 8; off >= 1; off >>= 1) rs += __shfl_xor(rs, off);
            l[i] = l[i] * alpha[i] + rs;
            m[i] = mnew;
            s0[i] = p0; s1[i] = p1;
        }
#pragma unroll
        for (int i = 0; i < 8; ++i) {
            o0[i] *= alpha[i]; o1[i] *= alpha[i];
            o2[i] *= alpha[i]; o3[i] *= alpha[i];
        }

#pragma unroll
        for (int i = 0; i < 8; ++i) {
            int row = half ? i + 8 : i;
            pbuf[row][laneRow]      = (bf16_t)s0[i];
            pbuf[row][16 + laneRow] = (bf16_t)s1[i];
        }
        __syncthreads();

        v16bf pa = afrag_bf16(&pbuf[laneRow][0], baseSel);
        o0 = wmma_bf16(pa, bfrag_bf16(&vt[0  + laneRow][half * 16]), o0);
        o1 = wmma_bf16(pa, bfrag_bf16(&vt[16 + laneRow][half * 16]), o1);
        o2 = wmma_bf16(pa, bfrag_bf16(&vt[32 + laneRow][half * 16]), o2);
        o3 = wmma_bf16(pa, bfrag_bf16(&vt[48 + laneRow][half * 16]), o3);
    }

#pragma unroll
    for (int t = 0; t < 4; ++t) {
        v8f o = (t == 0) ? o0 : (t == 1) ? o1 : (t == 2) ? o2 : o3;
#pragma unroll
        for (int i = 0; i < 8; ++i) {
            int n = qblk * 16 + (half ? i + 8 : i);
            int c = h * 64 + t * 16 + laneRow;
            float v = o[i] / l[i];
            x[((size_t)(n * 4 + b)) * 1024 + c] = (bf16_t)v;
        }
    }
}

// ---------------------------------------------------------------------------
// Output projection: d_out = x @ out_w^T + out_b  (bf16 A/W, fp32 out).
// Same 32x64 wave tiling as the QKV projection.
// ---------------------------------------------------------------------------
__global__ __launch_bounds__(256) void out_proj_kernel(
    const bf16_t* __restrict__ X, const bf16_t* __restrict__ W,
    const float* __restrict__ bias, float* __restrict__ Out)
{
    const int lane = threadIdx.x, wv = threadIdx.y;
    const int laneRow = lane & 15, half = lane >> 4;
    const int baseSel = half * 8, koff = half * 16;
    const int rBase = blockIdx.x * 128 + (wv & 3) * 32;
    const int nBase = blockIdx.y * 128 + (wv >> 2) * 64;

    v8f acc[2][4];
#pragma unroll
    for (int tr = 0; tr < 2; ++tr)
#pragma unroll
        for (int tc = 0; tc < 4; ++tc) acc[tr][tc] = {};

    const bf16_t* a0p = X + (size_t)(rBase + laneRow) * 1024;
    const bf16_t* a1p = X + (size_t)(rBase + 16 + laneRow) * 1024;
    const bf16_t* bp[4];
#pragma unroll
    for (int t = 0; t < 4; ++t)
        bp[t] = W + (size_t)(nBase + t * 16 + laneRow) * 1024 + koff;

    for (int kk = 0; kk < 1024; kk += 32) {
        v16bf a0 = afrag_bf16(a0p + kk, baseSel);
        v16bf a1 = afrag_bf16(a1p + kk, baseSel);
#pragma unroll
        for (int t = 0; t < 4; ++t) {
            v16bf b = bfrag_bf16(bp[t] + kk);
            acc[0][t] = wmma_bf16(a0, b, acc[0][t]);
            acc[1][t] = wmma_bf16(a1, b, acc[1][t]);
        }
    }

#pragma unroll
    for (int tr = 0; tr < 2; ++tr)
#pragma unroll
        for (int tc = 0; tc < 4; ++tc)
#pragma unroll
            for (int i = 0; i < 8; ++i) {
                int r = rBase + tr * 16 + (half ? i + 8 : i);
                int c = nBase + tc * 16 + laneRow;
                Out[(size_t)r * 1024 + c] = acc[tr][tc][i] + bias[c];
            }
}

extern "C" void kernel_launch(void* const* d_in, const int* in_sizes, int n_in,
                              void* d_out, int out_size, void* d_ws, size_t ws_size,
                              hipStream_t stream) {
    const float* query       = (const float*)d_in[0];
    const float* key         = (const float*)d_in[1];
    const float* value       = (const float*)d_in[2];
    const float* in_proj_w   = (const float*)d_in[3];
    const float* in_proj_b   = (const float*)d_in[4];
    const float* logit_scale = (const float*)d_in[5];
    const float* out_w       = (const float*)d_in[6];
    const float* out_b       = (const float*)d_in[7];

    const size_t M4 = (size_t)4 * 1024 * 1024;   // 4M elems
    bf16_t* qc    = (bf16_t*)d_ws;               // bf16 copies of inputs
    bf16_t* kc    = qc + M4;
    bf16_t* vc    = kc + M4;
    bf16_t* wqkvc = vc + M4;                     // 3M elems
    bf16_t* woutc = wqkvc + 3 * 1024 * 1024;     // 1M elems
    bf16_t* qb    = woutc + 1024 * 1024;         // head-major projections
    bf16_t* kb    = qb + M4;
    bf16_t* vb    = kb + M4;
    bf16_t* xb    = vb + M4;                     // attention output (n,b,c)

    // One-pass fp32 -> bf16 conversions (all compute-bound work then runs bf16).
    cvt_f32_bf16_kernel<<<4096, 256, 0, stream>>>(query, qc);
    cvt_f32_bf16_kernel<<<4096, 256, 0, stream>>>(key, kc);
    cvt_f32_bf16_kernel<<<4096, 256, 0, stream>>>(value, vc);
    cvt_f32_bf16_kernel<<<3072, 256, 0, stream>>>(in_proj_w, wqkvc);
    cvt_f32_bf16_kernel<<<1024, 256, 0, stream>>>(out_w, woutc);

    dim3 gemmBlk(32, 8);
    // QKV projections: M=4096, N=1024 per z; block tile 128x128.
    qkv_proj_kernel<<<dim3(32, 8, 3), gemmBlk, 0, stream>>>(
        qc, kc, vc, wqkvc, in_proj_b, qb, kb, vb);

    l2norm_kernel<<<dim3(1024, 64, 2), 32, 0, stream>>>(qb, kb);

    attn_kernel<<<dim3(64, 64), 32, 0, stream>>>(qb, kb, vb, logit_scale, xb);

    out_proj_kernel<<<dim3(32, 8), gemmBlk, 0, stream>>>(
        xb, woutc, out_b, (float*)d_out);
}
